// FramePIGNN_85873576116399
// MI455X (gfx1250) — compile-verified, matrix operand order checked by reading, not run on hardware
//
#include <hip/hip_runtime.h>
#include <hip/hip_bf16.h>

typedef __attribute__((ext_vector_type(16))) _Float16 v16h;
typedef __attribute__((ext_vector_type(8)))  float    v8f;

// silu via fast hardware reciprocal: v_exp_f32 + v_rcp_f32 + v_mul (no IEEE div expansion)
__device__ __forceinline__ float silu_f(float x) {
  return x * __builtin_amdgcn_rcpf(1.0f + __expf(-x));
}

__device__ __forceinline__ v8f wmma_fp16(v16h a, v16h b, v8f c) {
  // D = A(16x32 f16) * B(32x16 f16) + C(16x16 f32)
  return __builtin_amdgcn_wmma_f32_16x16x32_f16(false, a, false, b, (short)0, c, false, false);
}

// A fragment: LDS row-major [16][ldk] f16.  lane L: M = L&15, K = kbase + (L>>4)*16 + j
__device__ __forceinline__ v16h frag_a(const _Float16* lds, int ldk, int lane, int kbase) {
  return *(const v16h*)(lds + (lane & 15) * ldk + kbase + ((lane >> 4) << 4));
}

// B fragment from packed weights: tiles [tn][tk][lane][16] (lane-contiguous 32B)
__device__ __forceinline__ v16h frag_b(const _Float16* pk, int Tk, int tn, int tk, int lane) {
  return *(const v16h*)(pk + (((size_t)tn * Tk + tk) << 9) + (lane << 4));
}

// ---------------------------------------------------------------------------
// Weight packer: f32 W[(k+row_off)*ldW + n] -> f16 WMMA B-fragment layout.
// lane L: n = tn*16 + (L&15), k = tk*32 + (L>>4)*16 + j, zero-pad k >= K.
// ---------------------------------------------------------------------------
__global__ void pack_w_kernel(const float* __restrict__ W, int row_off, int K,
                              int Nn, int ldW, _Float16* __restrict__ out) {
  int Tk = (K + 31) >> 5;
  int total = (Nn >> 4) * Tk * 512;
  for (int idx = blockIdx.x * blockDim.x + threadIdx.x; idx < total;
       idx += gridDim.x * blockDim.x) {
    int tn   = idx / (Tk * 512);
    int rem  = idx - tn * Tk * 512;
    int tk   = rem >> 9;
    int r2   = rem & 511;
    int lane = r2 >> 4;
    int j    = r2 & 15;
    int n = (tn << 4) + (lane & 15);
    int k = (tk << 5) + ((lane >> 4) << 4) + j;
    float v = (k < K) ? W[(size_t)(k + row_off) * ldW + n] : 0.0f;
    out[idx] = (_Float16)v;
  }
}

// ---------------------------------------------------------------------------
// Encoder: h = silu(silu(x @ W1 + b1) @ W2 + b2), x: N x 9 (K padded to 32)
// 256 threads = 8 waves; block handles 16 nodes; wave w -> cols [16w,16w+16)
// ---------------------------------------------------------------------------
__global__ __launch_bounds__(256) void enc_kernel(
    const float* __restrict__ x, const float* __restrict__ b1,
    const float* __restrict__ b2, const _Float16* __restrict__ W1p,
    const _Float16* __restrict__ W2p, float* __restrict__ h32,
    _Float16* __restrict__ h16, int N) {
  __shared__ __align__(32) _Float16 Ax[16 * 32];
  __shared__ __align__(32) _Float16 H1[16 * 128];
  int t = threadIdx.x, lane = t & 31, w = t >> 5;
  int n0 = blockIdx.x * 16;
  {
    int m = t >> 4, c0 = (t & 15) * 2;
    int row = min(n0 + m, N - 1);
#pragma unroll
    for (int u = 0; u < 2; ++u) {
      int c = c0 + u;
      float v = (c < 9) ? x[(size_t)row * 9 + c] : 0.0f;
      Ax[m * 32 + c] = (_Float16)v;
    }
  }
  __syncthreads();
  v8f acc = {};
  acc = wmma_fp16(frag_a(Ax, 32, lane, 0), frag_b(W1p, 1, w, 0, lane), acc);
  int n = (lane & 15) + w * 16;
  {
    float bv = b1[n];
#pragma unroll
    for (int r = 0; r < 8; ++r) {
      int m = r + ((lane >> 4) << 3);
      H1[m * 128 + n] = (_Float16)silu_f(acc[r] + bv);
    }
  }
  __syncthreads();
  v8f acc2 = {};
#pragma unroll
  for (int tk = 0; tk < 4; ++tk)
    acc2 = wmma_fp16(frag_a(H1, 128, lane, tk * 32), frag_b(W2p, 4, w, tk, lane), acc2);
  {
    float bv = b2[n];
#pragma unroll
    for (int r = 0; r < 8; ++r) {
      int m = r + ((lane >> 4) << 3);
      int row = n0 + m;
      if (row < N) {
        float v = silu_f(acc2[r] + bv);
        h32[(size_t)row * 128 + n] = v;
        h16[(size_t)row * 128 + n] = (_Float16)v;
      }
    }
  }
}

// ---------------------------------------------------------------------------
// Message MLP + scatter:  m = silu([h_src|h_dst|ea] @ W1 + b1) @ W2 + b2
// K split 128(src) + 128(dst) + 32(ea padded).  atomicAdd into agg[dst].
// 256 threads = 8 waves; 32 edges per block (two 16-row A-tiles per wave,
// B fragment held in registers and reused across both -> 26 WMMAs/wave).
// ---------------------------------------------------------------------------
__global__ __launch_bounds__(256) void msg_kernel(
    const _Float16* __restrict__ h16, const int* __restrict__ ei,
    const float* __restrict__ ea, const float* __restrict__ b1,
    const float* __restrict__ b2, const _Float16* __restrict__ W1p,
    const _Float16* __restrict__ W2p, float* __restrict__ agg, int E) {
  __shared__ __align__(32) _Float16 As[32 * 128];
  __shared__ __align__(32) _Float16 Ad[32 * 128];
  __shared__ __align__(32) _Float16 Ae[32 * 32];
  __shared__ __align__(32) _Float16 M1[32 * 128];
  __shared__ int sdst[32];
  int t = threadIdx.x, lane = t & 31, w = t >> 5;
  int e0 = blockIdx.x * 32;
  {
    int c8 = t & 15;
#pragma unroll
    for (int rr = 0; rr < 2; ++rr) {
      int m = (t >> 4) + rr * 16;
      int e = min(e0 + m, E - 1);
      int s = ei[e], d = ei[E + e];
      if (c8 == 0) sdst[m] = d;
      *(uint4*)(As + m * 128 + c8 * 8) = *(const uint4*)(h16 + (size_t)s * 128 + c8 * 8);
      *(uint4*)(Ad + m * 128 + c8 * 8) = *(const uint4*)(h16 + (size_t)d * 128 + c8 * 8);
      int c0 = c8 * 2;
#pragma unroll
      for (int u = 0; u < 2; ++u) {
        int c = c0 + u;
        float v = (c < 11) ? ea[(size_t)e * 11 + c] : 0.0f;
        Ae[m * 32 + c] = (_Float16)v;
      }
    }
  }
  __syncthreads();
  v8f acc0 = {}, acc1 = {};
#pragma unroll
  for (int tk = 0; tk < 4; ++tk) {
    v16h b = frag_b(W1p, 9, w, tk, lane);
    acc0 = wmma_fp16(frag_a(As, 128, lane, tk * 32), b, acc0);
    acc1 = wmma_fp16(frag_a(As + 16 * 128, 128, lane, tk * 32), b, acc1);
  }
#pragma unroll
  for (int tk = 0; tk < 4; ++tk) {
    v16h b = frag_b(W1p, 9, w, 4 + tk, lane);
    acc0 = wmma_fp16(frag_a(Ad, 128, lane, tk * 32), b, acc0);
    acc1 = wmma_fp16(frag_a(Ad + 16 * 128, 128, lane, tk * 32), b, acc1);
  }
  {
    v16h b = frag_b(W1p, 9, w, 8, lane);
    acc0 = wmma_fp16(frag_a(Ae, 32, lane, 0), b, acc0);
    acc1 = wmma_fp16(frag_a(Ae + 16 * 32, 32, lane, 0), b, acc1);
  }
  int n = (lane & 15) + w * 16;
  {
    float bv = b1[n];
#pragma unroll
    for (int r = 0; r < 8; ++r) {
      int m = r + ((lane >> 4) << 3);
      M1[m * 128 + n]        = (_Float16)silu_f(acc0[r] + bv);
      M1[(m + 16) * 128 + n] = (_Float16)silu_f(acc1[r] + bv);
    }
  }
  __syncthreads();
  v8f c0 = {}, c1 = {};
#pragma unroll
  for (int tk = 0; tk < 4; ++tk) {
    v16h b = frag_b(W2p, 4, w, tk, lane);
    c0 = wmma_fp16(frag_a(M1, 128, lane, tk * 32), b, c0);
    c1 = wmma_fp16(frag_a(M1 + 16 * 128, 128, lane, tk * 32), b, c1);
  }
  {
    float bv = b2[n];
#pragma unroll
    for (int r = 0; r < 8; ++r) {
      int m = r + ((lane >> 4) << 3);
      if (e0 + m < E)
        atomicAdd(&agg[(size_t)sdst[m] * 128 + n], c0[r] + bv);
      if (e0 + m + 16 < E)
        atomicAdd(&agg[(size_t)sdst[m + 16] * 128 + n], c1[r] + bv);
    }
  }
}

// ---------------------------------------------------------------------------
// Update: h = silu([h|agg] @ updW + b + h).  K = 256 -> 8 WMMAs per wave.
// ---------------------------------------------------------------------------
__global__ __launch_bounds__(256) void upd_kernel(
    const float* __restrict__ agg, const float* __restrict__ ub,
    const _Float16* __restrict__ Wp, float* __restrict__ h32,
    _Float16* __restrict__ h16, int N) {
  __shared__ __align__(32) _Float16 A[16 * 256];
  int t = threadIdx.x, lane = t & 31, w = t >> 5;
  int n0 = blockIdx.x * 16;
  {
    int m = t >> 4, chunk = t & 15;
    int row = min(n0 + m, N - 1);
    if (chunk < 8) {
      const uint4* s = (const uint4*)(h16 + (size_t)row * 128 + chunk * 16);
      uint4* d = (uint4*)(A + m * 256 + chunk * 16);
      d[0] = s[0]; d[1] = s[1];
    } else {
      int cb = (chunk - 8) * 16;
#pragma unroll
      for (int u = 0; u < 16; ++u)
        A[m * 256 + 128 + cb + u] = (_Float16)agg[(size_t)row * 128 + cb + u];
    }
  }
  __syncthreads();
  v8f acc = {};
#pragma unroll
  for (int tk = 0; tk < 8; ++tk)
    acc = wmma_fp16(frag_a(A, 256, lane, tk * 32), frag_b(Wp, 8, w, tk, lane), acc);
  int n = (lane & 15) + w * 16;
  float bv = ub[n];
#pragma unroll
  for (int r = 0; r < 8; ++r) {
    int m = r + ((lane >> 4) << 3);
    int row = n0 + m;
    if (row < N) {
      float v = silu_f(acc[r] + bv + h32[(size_t)row * 128 + n]);
      h32[(size_t)row * 128 + n] = v;
      h16[(size_t)row * 128 + n] = (_Float16)v;
    }
  }
}

// ---------------------------------------------------------------------------
// Decoder stage 1: base[c] = h[n1] @ efA(128x64) + h[n2] @ efB(128x64) + b1
// (ef_W1 rows 1..128 and 129..256; shared across all 5 quad points)
// 128 threads = 4 waves; 16 pairs per block; wave w -> cols [16w,16w+16)
// ---------------------------------------------------------------------------
__global__ __launch_bounds__(128) void ef1_kernel(
    const _Float16* __restrict__ h16, const int* __restrict__ conn,
    const float* __restrict__ b1, const _Float16* __restrict__ Ap,
    const _Float16* __restrict__ Bp, float* __restrict__ base, int EC) {
  __shared__ __align__(32) _Float16 A1[16 * 128];
  __shared__ __align__(32) _Float16 A2[16 * 128];
  int t = threadIdx.x, lane = t & 31, w = t >> 5;
  int c0 = blockIdx.x * 16;
  {
    int m = t >> 3, q = t & 7;
    int c = min(c0 + m, EC - 1);
    int n1 = conn[(size_t)c * 2], n2 = conn[(size_t)c * 2 + 1];
    const uint4* s1 = (const uint4*)(h16 + (size_t)n1 * 128 + q * 16);
    const uint4* s2 = (const uint4*)(h16 + (size_t)n2 * 128 + q * 16);
    uint4* d1 = (uint4*)(A1 + m * 128 + q * 16);
    uint4* d2 = (uint4*)(A2 + m * 128 + q * 16);
    d1[0] = s1[0]; d1[1] = s1[1];
    d2[0] = s2[0]; d2[1] = s2[1];
  }
  __syncthreads();
  v8f acc = {};
#pragma unroll
  for (int tk = 0; tk < 4; ++tk)
    acc = wmma_fp16(frag_a(A1, 128, lane, tk * 32), frag_b(Ap, 4, w, tk, lane), acc);
#pragma unroll
  for (int tk = 0; tk < 4; ++tk)
    acc = wmma_fp16(frag_a(A2, 128, lane, tk * 32), frag_b(Bp, 4, w, tk, lane), acc);
  int n = (lane & 15) + w * 16;
  float bv = b1[n];
#pragma unroll
  for (int r = 0; r < 8; ++r) {
    int m = r + ((lane >> 4) << 3);
    int c = c0 + m;
    if (c < EC) base[(size_t)c * 64 + n] = acc[r] + bv;
  }
}

// ---------------------------------------------------------------------------
// Decoder stage 2: per (pair, quad): f1 = silu(base + xi*w0);
// f2 = silu(f1 @ W2 + b2); fields = (f2 @ W3 + b3) * mask.  Rows = EC*5.
// ---------------------------------------------------------------------------
__global__ __launch_bounds__(128) void ef2_kernel(
    const float* __restrict__ base, const int* __restrict__ conn,
    const float* __restrict__ bc, const float* __restrict__ w0,
    const float* __restrict__ b2, const _Float16* __restrict__ W2p,
    const float* __restrict__ W3, const float* __restrict__ b3,
    float* __restrict__ out_xi, float* __restrict__ out_fields, int EC) {
  __shared__ __align__(32) _Float16 F1[16 * 64];
  __shared__ __align__(32) _Float16 F2[16 * 64];
  int t = threadIdx.x, lane = t & 31, w = t >> 5;
  int g0 = blockIdx.x * 16;
  int R = EC * 5;
  {
    int m = t >> 3, kb = (t & 7) * 8;
    int g = min(g0 + m, R - 1);
    int c = g / 5, q = g - c * 5;
    float xi = 0.25f * (float)q;
#pragma unroll
    for (int u = 0; u < 8; ++u) {
      int k = kb + u;
      F1[m * 64 + k] = (_Float16)silu_f(base[(size_t)c * 64 + k] + xi * w0[k]);
    }
  }
  __syncthreads();
  v8f acc = {};
  acc = wmma_fp16(frag_a(F1, 64, lane, 0),  frag_b(W2p, 2, w, 0, lane), acc);
  acc = wmma_fp16(frag_a(F1, 64, lane, 32), frag_b(W2p, 2, w, 1, lane), acc);
  int n = (lane & 15) + w * 16;
  float bv = b2[n];
#pragma unroll
  for (int r = 0; r < 8; ++r) {
    int m = r + ((lane >> 4) << 3);
    F2[m * 64 + n] = (_Float16)silu_f(acc[r] + bv);
  }
  __syncthreads();
  if (t < 48) {
    int m = t / 3, j = t - m * 3;
    int g = g0 + m;
    if (g < R) {
      int c = g / 5, q = g - c * 5;
      float xi = 0.25f * (float)q;
      float dot = b3[j];
#pragma unroll
      for (int k = 0; k < 64; ++k) dot += (float)F2[m * 64 + k] * W3[k * 3 + j];
      int n1 = conn[(size_t)c * 2], n2 = conn[(size_t)c * 2 + 1];
      float mask = (1.0f - bc[n1] * (1.0f - xi)) * (1.0f - bc[n2] * xi);
      out_fields[(size_t)g * 3 + j] = dot * mask;
    }
  }
  if (t < 16) {
    int g = g0 + t;
    if (g < R) out_xi[g] = 0.25f * (float)(g % 5);
  }
}

// ---------------------------------------------------------------------------
// Output assembly: h (f32) and prop_I22 pass-through
// ---------------------------------------------------------------------------
__global__ void copy_out_kernel(const float* __restrict__ h32,
                                const float* __restrict__ prop,
                                float* __restrict__ out_h,
                                float* __restrict__ out_prop, int nh, int np) {
  int stride = gridDim.x * blockDim.x;
  for (int k = blockIdx.x * blockDim.x + threadIdx.x; k < nh; k += stride)
    out_h[k] = h32[k];
  for (int k = blockIdx.x * blockDim.x + threadIdx.x; k < np; k += stride)
    out_prop[k] = prop[k];
}

extern "C" void kernel_launch(void* const* d_in, const int* in_sizes, int n_in,
                              void* d_out, int out_size, void* d_ws, size_t ws_size,
                              hipStream_t stream) {
  const float* x     = (const float*)d_in[0];
  const int*   ei    = (const int*)d_in[1];
  const float* ea    = (const float*)d_in[2];
  const int*   conn  = (const int*)d_in[3];
  const float* bc    = (const float*)d_in[4];
  const float* prop  = (const float*)d_in[5];
  const float* encW1 = (const float*)d_in[7];
  const float* encb1 = (const float*)d_in[8];
  const float* encW2 = (const float*)d_in[9];
  const float* encb2 = (const float*)d_in[10];
  const float* msgW1 = (const float*)d_in[11];
  const float* msgb1 = (const float*)d_in[12];
  const float* msgW2 = (const float*)d_in[13];
  const float* msgb2 = (const float*)d_in[14];
  const float* updW  = (const float*)d_in[15];
  const float* updb  = (const float*)d_in[16];
  const float* efW1  = (const float*)d_in[17];
  const float* efb1  = (const float*)d_in[18];
  const float* efW2  = (const float*)d_in[19];
  const float* efb2  = (const float*)d_in[20];
  const float* efW3  = (const float*)d_in[21];
  const float* efb3  = (const float*)d_in[22];

  const int N  = in_sizes[0] / 9;
  const int E  = in_sizes[1] / 2;
  const int EC = in_sizes[3] / 2;
  const int NL = 6;

  // ---- workspace carve (256B aligned) ----
  size_t off = 0;
  auto carve = [&](size_t bytes) -> void* {
    void* p = (char*)d_ws + off;
    off += (bytes + 255) & ~(size_t)255;
    return p;
  };
  float*     h32    = (float*)carve((size_t)N * 128 * 4);
  _Float16*  h16    = (_Float16*)carve((size_t)N * 128 * 2);
  float*     agg    = (float*)carve((size_t)N * 128 * 4);
  float*     basef  = (float*)carve((size_t)EC * 64 * 4);
  _Float16*  encW1p = (_Float16*)carve(1 * 8 * 512 * 2);
  _Float16*  encW2p = (_Float16*)carve(4 * 8 * 512 * 2);
  _Float16*  msgW1p = (_Float16*)carve((size_t)NL * 9 * 8 * 512 * 2);
  _Float16*  msgW2p = (_Float16*)carve((size_t)NL * 4 * 8 * 512 * 2);
  _Float16*  updWp  = (_Float16*)carve((size_t)NL * 8 * 8 * 512 * 2);
  _Float16*  efAp   = (_Float16*)carve(4 * 4 * 512 * 2);
  _Float16*  efBp   = (_Float16*)carve(4 * 4 * 512 * 2);
  _Float16*  efW2p  = (_Float16*)carve(2 * 4 * 512 * 2);
  (void)ws_size;

  auto pack = [&](const float* W, int row_off, int K, int Nn, int ldW, _Float16* out) {
    int Tk = (K + 31) >> 5;
    int total = (Nn >> 4) * Tk * 512;
    pack_w_kernel<<<(total + 255) / 256, 256, 0, stream>>>(W, row_off, K, Nn, ldW, out);
  };

  // ---- weight repack (f32 -> f16 WMMA B-fragment layout) ----
  pack(encW1, 0, 9, 128, 128, encW1p);
  pack(encW2, 0, 128, 128, 128, encW2p);
  for (int l = 0; l < NL; ++l) {
    pack(msgW1 + (size_t)l * 267 * 128, 0, 267, 128, 128, msgW1p + (size_t)l * 9 * 8 * 512);
    pack(msgW2 + (size_t)l * 128 * 128, 0, 128, 128, 128, msgW2p + (size_t)l * 4 * 8 * 512);
    pack(updW  + (size_t)l * 256 * 128, 0, 256, 128, 128, updWp  + (size_t)l * 8 * 8 * 512);
  }
  pack(efW1, 1,   128, 64, 64, efAp);
  pack(efW1, 129, 128, 64, 64, efBp);
  pack(efW2, 0,   64,  64, 64, efW2p);

  // ---- encoder ----
  enc_kernel<<<(N + 15) / 16, 256, 0, stream>>>(x, encb1, encb2, encW1p, encW2p,
                                                h32, h16, N);

  // ---- 6 message-passing layers ----
  for (int l = 0; l < NL; ++l) {
    hipMemsetAsync(agg, 0, (size_t)N * 128 * 4, stream);
    msg_kernel<<<(E + 31) / 32, 256, 0, stream>>>(
        h16, ei, ea, msgb1 + l * 128, msgb2 + l * 128,
        msgW1p + (size_t)l * 9 * 8 * 512, msgW2p + (size_t)l * 4 * 8 * 512, agg, E);
    upd_kernel<<<(N + 15) / 16, 256, 0, stream>>>(
        agg, updb + l * 128, updWp + (size_t)l * 8 * 8 * 512, h32, h16, N);
  }

  // ---- edge-field decoder ----
  ef1_kernel<<<(EC + 15) / 16, 128, 0, stream>>>(h16, conn, efb1, efAp, efBp,
                                                 basef, EC);
  const int nh = N * 128;
  float* out_xi     = (float*)d_out + nh;
  float* out_fields = (float*)d_out + nh + EC * 5;
  float* out_prop   = (float*)d_out + nh + EC * 5 + EC * 5 * 3;
  ef2_kernel<<<(EC * 5 + 15) / 16, 128, 0, stream>>>(
      basef, conn, bc, efW1 /* row 0 = xi weights */, efb2, efW2p, efW3, efb3,
      out_xi, out_fields, EC);

  // ---- outputs: h and prop_I22 ----
  copy_out_kernel<<<2048, 256, 0, stream>>>(h32, prop, (float*)d_out, out_prop,
                                            nh, EC);
}
